// CNLinkPredictor_39556648796289
// MI455X (gfx1250) — matrix operand order repaired and checked by reading, not compiled, not run on hardware
//
#include <hip/hip_runtime.h>
#include <hip/hip_bf16.h>
#include <cstdint>
#include <cstddef>

// Problem constants (reference: N=8192 nodes, E=8192 edges, C=256)
#define C_DIM   256
#define N_NODES 8192
#define N_EDGES 8192

typedef __bf16 bf16_t;
typedef bf16_t v16bf __attribute__((ext_vector_type(16)));
typedef bf16_t v8bf  __attribute__((ext_vector_type(8)));
typedef float  v8f   __attribute__((ext_vector_type(8)));

__device__ __forceinline__ bf16_t f2bf(float f) {
  union { float f; uint32_t u; } v; v.f = f;
  uint32_t r = v.u + 0x7FFFu + ((v.u >> 16) & 1u);   // round-to-nearest-even
  return __builtin_bit_cast(bf16_t, (uint16_t)(r >> 16));
}
__device__ __forceinline__ float bf2f(bf16_t b) {
  uint32_t u = ((uint32_t)__builtin_bit_cast(uint16_t, b)) << 16;
  return __builtin_bit_cast(float, u);
}

union FragBF { v16bf v; v8bf h[2]; };

// CDNA5 async copy: 16 bytes global -> LDS, tracked by ASYNCcnt.
__device__ __forceinline__ void async_ld16(uint32_t lds_off, const void* gaddr) {
  asm volatile("global_load_async_to_lds_b128 %0, %1, off"
               :: "v"(lds_off), "v"(gaddr) : "memory");
}
__device__ __forceinline__ void wait_async0() {
  asm volatile("s_wait_asynccnt 0" ::: "memory");
}

// -----------------------------------------------------------------------------
// fp32 -> bf16 flat conversion (n multiple of 2048)
// -----------------------------------------------------------------------------
__global__ __launch_bounds__(256)
void cvt_f32_bf16(const float* __restrict__ in, bf16_t* __restrict__ out, int n) {
  int i0 = (blockIdx.x * 256 + threadIdx.x) * 8;
  if (i0 >= n) return;
  v8bf v;
  for (int q = 0; q < 8; ++q) v[q] = f2bf(in[i0 + q]);
  *(v8bf*)(out + i0) = v;
}

// -----------------------------------------------------------------------------
// Weight convert + transpose: Wt[n][k] = bf16(W[k][n]), W is 256x256
// 32x32 tile per block, grid (8,8), 256 threads (32x8)
// -----------------------------------------------------------------------------
__global__ __launch_bounds__(256)
void cvt_w_transpose(const float* __restrict__ W, bf16_t* __restrict__ Wt) {
  __shared__ float t[32][33];
  const int bx = blockIdx.x * 32, by = blockIdx.y * 32;
  const int lx = threadIdx.x & 31, ly = threadIdx.x >> 5;
  for (int r = ly; r < 32; r += 8)
    t[r][lx] = W[(size_t)(by + r) * C_DIM + bx + lx];
  __syncthreads();
  for (int r = ly; r < 32; r += 8)
    Wt[(size_t)(bx + r) * C_DIM + by + lx] = f2bf(t[lx][r]);
}

// -----------------------------------------------------------------------------
// bf16 WMMA GEMM: out = epilogue(A[M x 256] @ W[256 x 256] + bias), W given
// pre-transposed as Wt[n][k]. Block tile 128(M) x 64(N), BK=32, 8 waves.
// Double-buffered LDS staged with async global->LDS copies (ASYNCcnt).
//   mode 0: relu(acc + bias)
//   mode 1: extra + relu(acc + bias)            (residual:  h = x + relu(...))
//   mode 2: relu(acc + bias) * beta + extra     (z = xcn*beta + xij)
// -----------------------------------------------------------------------------
__global__ __launch_bounds__(256)
void gemm_bf16_wmma(const bf16_t* __restrict__ A, const bf16_t* __restrict__ Wt,
                    const float* __restrict__ bias, bf16_t* __restrict__ out,
                    const bf16_t* __restrict__ extra, const float* __restrict__ beta_ptr,
                    int mode) {
  __shared__ __align__(16) bf16_t As[2][128 * 32];   // [m][k], 8 KB each
  __shared__ __align__(16) bf16_t Bs[2][64 * 32];    // [n][k], 4 KB each

  const int tid  = threadIdx.x;
  const int lane = tid & 31;
  const int wave = tid >> 5;
  const int m0 = blockIdx.x * 128;
  const int n0 = blockIdx.y * 64;

  v8f acc[4];
  for (int nt = 0; nt < 4; ++nt)
    for (int q = 0; q < 8; ++q) acc[nt][q] = 0.0f;

  const int frow = lane & 15;   // M row (A) / N col (B,C,D) within 16x16 tile
  const int kh   = lane >> 4;   // K-half select per ISA 16-bit fragment layouts

  // Stage one BK=32 K-slab into LDS buffer `sel` via async copies.
  auto stage = [&](int k0, int sel) {
    const uint32_t asb = (uint32_t)(uintptr_t)(void*)&As[sel][0];
    const uint32_t bsb = (uint32_t)(uintptr_t)(void*)&Bs[sel][0];
    // A tile: 128 rows x 32 bf16 (64 B/row) = 512 x 16B chunks, 2 per thread
    for (int it = 0; it < 2; ++it) {
      int c = it * 256 + tid;
      int r = c >> 2, part = c & 3;
      async_ld16(asb + (uint32_t)(r * 64 + part * 16),
                 A + (size_t)(m0 + r) * C_DIM + k0 + part * 8);
    }
    // B tile: 64 rows x 32 bf16 = 256 x 16B chunks, 1 per thread
    {
      int n = tid >> 2, part = tid & 3;
      async_ld16(bsb + (uint32_t)(n * 64 + part * 16),
                 Wt + (size_t)(n0 + n) * C_DIM + k0 + part * 8);
    }
  };

  stage(0, 0);
#pragma unroll
  for (int kb = 0; kb < 8; ++kb) {
    const int sel = kb & 1;
    wait_async0();
    __syncthreads();                 // staged slab visible; prev compute done
    if (kb + 1 < 8) stage((kb + 1) * 32, sel ^ 1);

    // A fragment (16x32): lane row = frow, K = kh*8+0..7 | 16+kh*8+0..7
    FragBF fa;
    const bf16_t* ap = &As[sel][(wave * 16 + frow) * 32];
    fa.h[0] = *(const v8bf*)(ap + kh * 8);
    fa.h[1] = *(const v8bf*)(ap + 16 + kh * 8);
#pragma unroll
    for (int nt = 0; nt < 4; ++nt) {
      // B fragment (32x16): lane col = frow, K = kh*8+0..7 | 16+kh*8+0..7
      FragBF fb;
      const bf16_t* bp = &Bs[sel][(nt * 16 + frow) * 32];
      fb.h[0] = *(const v8bf*)(bp + kh * 8);
      fb.h[1] = *(const v8bf*)(bp + 16 + kh * 8);
      acc[nt] = __builtin_amdgcn_wmma_f32_16x16x32_bf16(
          false, fa.v, false, fb.v, (short)0, acc[nt], false, false);
    }
  }

  const float betav = (mode == 2) ? beta_ptr[0] : 0.0f;
  const int rbase = kh * 8;          // C/D layout: lanes 16-31 hold M = v+8
#pragma unroll
  for (int nt = 0; nt < 4; ++nt) {
    int gn = n0 + nt * 16 + frow;
    float bval = bias[gn];
#pragma unroll
    for (int v = 0; v < 8; ++v) {
      int gm = m0 + wave * 16 + rbase + v;
      size_t oi = (size_t)gm * C_DIM + gn;
      float val = fmaxf(acc[nt][v] + bval, 0.0f);
      if (mode == 1)      val = bf2f(extra[oi]) + val;
      else if (mode == 2) val = val * betav + bf2f(extra[oi]);
      out[oi] = f2bf(val);
    }
  }
}

// -----------------------------------------------------------------------------
// xi * xj gather (bf16 in/out, fp32 math)
// -----------------------------------------------------------------------------
__global__ __launch_bounds__(256)
void gather_mul(const bf16_t* __restrict__ xb, const int* __restrict__ ei,
                bf16_t* __restrict__ out) {
  int e = blockIdx.x, c = threadIdx.x;
  int i = ei[e], j = ei[N_EDGES + e];
  float v = bf2f(xb[(size_t)i * C_DIM + c]) * bf2f(xb[(size_t)j * C_DIM + c]);
  out[(size_t)e * C_DIM + c] = f2bf(v);
}

// -----------------------------------------------------------------------------
// Common-neighbor SpMM: out[e,:] = sum_k adj[i_e,k]*adj[j_e,k] * h[k,:]
// Streaming scan of two fp32 adjacency rows (bandwidth bound: 512 MB total),
// hit indices collected in LDS, then h rows (bf16) accumulated in fp32.
// -----------------------------------------------------------------------------
#define MAX_CN 1024
__global__ __launch_bounds__(256)
void spmm_cn(const float* __restrict__ adj, const int* __restrict__ ei,
             const bf16_t* __restrict__ h, bf16_t* __restrict__ out) {
  __shared__ int s_cnt;
  __shared__ int s_idx[MAX_CN];
  const int e = blockIdx.x, c = threadIdx.x;
  if (c == 0) s_cnt = 0;
  __syncthreads();
  const int i = ei[e], j = ei[N_EDGES + e];
  const float4* ai = (const float4*)(adj + (size_t)i * N_NODES);
  const float4* aj = (const float4*)(adj + (size_t)j * N_NODES);
  for (int k4 = c; k4 < N_NODES / 4; k4 += 256) {
    float4 a = ai[k4];
    float4 b = aj[k4];
    if (a.x != 0.f && b.x != 0.f) { int p = atomicAdd(&s_cnt, 1); if (p < MAX_CN) s_idx[p] = k4 * 4 + 0; }
    if (a.y != 0.f && b.y != 0.f) { int p = atomicAdd(&s_cnt, 1); if (p < MAX_CN) s_idx[p] = k4 * 4 + 1; }
    if (a.z != 0.f && b.z != 0.f) { int p = atomicAdd(&s_cnt, 1); if (p < MAX_CN) s_idx[p] = k4 * 4 + 2; }
    if (a.w != 0.f && b.w != 0.f) { int p = atomicAdd(&s_cnt, 1); if (p < MAX_CN) s_idx[p] = k4 * 4 + 3; }
  }
  __syncthreads();
  int n = s_cnt < MAX_CN ? s_cnt : MAX_CN;
  float acc = 0.0f;
  for (int t = 0; t < n; ++t) acc += bf2f(h[(size_t)s_idx[t] * C_DIM + c]);
  out[(size_t)e * C_DIM + c] = f2bf(acc);
}

// -----------------------------------------------------------------------------
// Final linear [E,256] @ [256,1] + b2 : one wave per edge, wave32 shuffle reduce
// -----------------------------------------------------------------------------
__global__ __launch_bounds__(256)
void final_lin(const bf16_t* __restrict__ t3, const float* __restrict__ w2,
               const float* __restrict__ b2, float* __restrict__ out) {
  const int lane = threadIdx.x & 31, wave = threadIdx.x >> 5;
  const int e = blockIdx.x * 8 + wave;
  const bf16_t* row = t3 + (size_t)e * C_DIM;
  float p = 0.0f;
  for (int q = 0; q < 8; ++q) { int cc = lane + q * 32; p += bf2f(row[cc]) * w2[cc]; }
  for (int off = 16; off > 0; off >>= 1) p += __shfl_down(p, off, 32);
  if (lane == 0) out[e] = p + b2[0];
}

// -----------------------------------------------------------------------------
extern "C" void kernel_launch(void* const* d_in, const int* in_sizes, int n_in,
                              void* d_out, int out_size, void* d_ws, size_t ws_size,
                              hipStream_t stream) {
  const float* x       = (const float*)d_in[0];
  const float* adj     = (const float*)d_in[1];
  const int*   ei      = (const int*)  d_in[2];
  const float* xlin_w1 = (const float*)d_in[3];
  const float* xlin_b1 = (const float*)d_in[4];
  const float* xlin_w2 = (const float*)d_in[5];
  const float* xlin_b2 = (const float*)d_in[6];
  const float* xcn_w1  = (const float*)d_in[7];
  const float* xcn_b1  = (const float*)d_in[8];
  const float* xcn_w2  = (const float*)d_in[9];
  const float* xcn_b2  = (const float*)d_in[10];
  const float* xij_w   = (const float*)d_in[11];
  const float* xij_b   = (const float*)d_in[12];
  const float* lin_w1  = (const float*)d_in[13];
  const float* lin_b1  = (const float*)d_in[14];
  const float* lin_w2  = (const float*)d_in[15];
  const float* lin_b2  = (const float*)d_in[16];
  const float* beta    = (const float*)d_in[17];
  float* out = (float*)d_out;

  const size_t NC = (size_t)N_NODES * C_DIM;      // 2M elements
  const size_t WW = (size_t)C_DIM * C_DIM;        // 64K elements
  bf16_t* p   = (bf16_t*)d_ws;
  bf16_t* xb  = p; p += NC;
  bf16_t* wt0 = p; p += WW;   // xlin_w1^T
  bf16_t* wt1 = p; p += WW;   // xlin_w2^T
  bf16_t* wt2 = p; p += WW;   // xij_w^T
  bf16_t* wt3 = p; p += WW;   // xcn_w1^T
  bf16_t* wt4 = p; p += WW;   // xcn_w2^T
  bf16_t* wt5 = p; p += WW;   // lin_w1^T
  bf16_t* buf0 = p; p += NC;
  bf16_t* buf1 = p; p += NC;
  bf16_t* buf2 = p; p += NC;
  if (ws_size < (size_t)((char*)p - (char*)d_ws)) return;   // ~17.6 MB scratch

  const dim3 wgrid(8, 8);
  cvt_f32_bf16 <<<(int)(NC / 8 / 256), 256, 0, stream>>>(x, xb, (int)NC);
  cvt_w_transpose<<<wgrid, 256, 0, stream>>>(xlin_w1, wt0);
  cvt_w_transpose<<<wgrid, 256, 0, stream>>>(xlin_w2, wt1);
  cvt_w_transpose<<<wgrid, 256, 0, stream>>>(xij_w,   wt2);
  cvt_w_transpose<<<wgrid, 256, 0, stream>>>(xcn_w1,  wt3);
  cvt_w_transpose<<<wgrid, 256, 0, stream>>>(xcn_w2,  wt4);
  cvt_w_transpose<<<wgrid, 256, 0, stream>>>(lin_w1,  wt5);

  const dim3 ggrid(N_NODES / 128, C_DIM / 64);    // 64 x 4 blocks
  // 1) t1 = relu(x @ xlin_w1 + b1)                        -> buf0
  gemm_bf16_wmma<<<ggrid, 256, 0, stream>>>(xb,   wt0, xlin_b1, buf0, nullptr, nullptr, 0);
  // 2) h = x + relu(t1 @ xlin_w2 + b2)                    -> buf1
  gemm_bf16_wmma<<<ggrid, 256, 0, stream>>>(buf0, wt1, xlin_b2, buf1, xb,      nullptr, 1);
  // 3) xixj[e] = x[i] * x[j]                              -> buf0
  gather_mul<<<N_EDGES, 256, 0, stream>>>(xb, ei, buf0);
  // 4) xij = relu(xixj @ xij_w + b)                       -> buf2
  gemm_bf16_wmma<<<ggrid, 256, 0, stream>>>(buf0, wt2, xij_b,   buf2, nullptr, nullptr, 0);
  // 5) xcn0[e] = (adj[i] ⊙ adj[j]) @ h                    -> buf0
  spmm_cn<<<N_EDGES, 256, 0, stream>>>(adj, ei, buf1, buf0);
  // 6) t2 = relu(xcn0 @ xcn_w1 + b1)                      -> buf1 (h dead)
  gemm_bf16_wmma<<<ggrid, 256, 0, stream>>>(buf0, wt3, xcn_b1,  buf1, nullptr, nullptr, 0);
  // 7) z = relu(t2 @ xcn_w2 + b2) * beta + xij            -> buf0
  gemm_bf16_wmma<<<ggrid, 256, 0, stream>>>(buf1, wt4, xcn_b2,  buf0, buf2,    beta,    2);
  // 8) t3 = relu(z @ lin_w1 + b1)                         -> buf1
  gemm_bf16_wmma<<<ggrid, 256, 0, stream>>>(buf0, wt5, lin_b1,  buf1, nullptr, nullptr, 0);
  // 9) out = t3 @ lin_w2 + b2                             -> d_out [E,1]
  final_lin<<<N_EDGES / 8, 256, 0, stream>>>(buf1, lin_w2, lin_b2, out);
}
